// Net_274877907102
// MI455X (gfx1250) — matrix operand order repaired and checked by reading, not compiled, not run on hardware
//
#include <hip/hip_runtime.h>
#include <hip/hip_bf16.h>

typedef __attribute__((ext_vector_type(16))) _Float16 v16h;
typedef __attribute__((ext_vector_type(8)))  float    v8f;
typedef int vi4 __attribute__((vector_size(16)));

#define BN_EPS 1e-5f

// ---------------------------------------------------------------------------
// WMMA 16x16x32 f16 fragment index helpers (CDNA5 ISA 7.12.2, wave32)
// slot = 2*vgpr + half
__device__ __forceinline__ int frag_k_a(int lane, int slot) {
    int p = slot >> 1, h = slot & 1;
    return ((p & 4) ? 16 : 0) + ((lane & 16) ? 8 : 0) + ((p & 3) * 2) + h;
}
__device__ __forceinline__ int frag_k_b(int lane, int slot) {
    int p = slot >> 1, h = slot & 1;
    return ((p & 4) ? 16 : 0) + ((lane & 16) ? 8 : 0) + ((p & 3) * 2) + h;
}

union AF  { v16h v; _Float16 h[16]; uint4 q[2]; };
union CF  { v8f  v; float     f[8]; };

// ---------------------------------------------------------------------------
// gfx1250 async global->LDS copy (16B per lane) + wait
__device__ __forceinline__ void async_copy16(const _Float16* g, _Float16* l) {
#if __has_builtin(__builtin_amdgcn_global_load_async_to_lds_b128)
    __builtin_amdgcn_global_load_async_to_lds_b128(
        (__attribute__((address_space(1))) vi4*)(_Float16*)g,
        (__attribute__((address_space(3))) vi4*)l,
        0, 0);
#else
    unsigned laddr = (unsigned)(unsigned long long)(__attribute__((address_space(3))) void*)l;
    asm volatile("global_load_async_to_lds_b128 %0, %1, off"
                 :: "v"(laddr), "v"((unsigned long long)g) : "memory");
#endif
}
__device__ __forceinline__ void async_wait0() {
#if __has_builtin(__builtin_amdgcn_s_wait_asynccnt)
    __builtin_amdgcn_s_wait_asynccnt(0);
#else
    asm volatile("s_wait_asynccnt 0x0" ::: "memory");
#endif
}

// ---------------------------------------------------------------------------
// Pre-pack conv weights [OC][K](f32) into per-lane A-fragment layout (f16).
// frag index = mt*KT + kt ; per frag: [lane][16 halves] contiguous (1 KB).
template<int OC, int K>
__global__ __launch_bounds__(32)
void pack_a_kernel(const float* __restrict__ w, _Float16* __restrict__ frags)
{
    constexpr int KP = (K + 31) & ~31, KT = KP / 32;
    const int frag = blockIdx.x;
    const int mt = frag / KT, kt = frag % KT;
    const int lane = threadIdx.x;
    const int mrow = mt * 16 + (lane & 15);
    _Float16* o = frags + ((size_t)frag * 32 + lane) * 16;
    #pragma unroll
    for (int slot = 0; slot < 16; ++slot) {
        const int k = kt * 32 + frag_k_a(lane, slot);
        o[slot] = (_Float16)((mrow < OC && k < K) ? w[(size_t)mrow * K + k] : 0.f);
    }
}

// Pre-pack FC weights W[K][50](f32) into B-fragment layout. frag = nt*KT + kt.
template<int K>
__global__ __launch_bounds__(32)
void pack_b_fc_kernel(const float* __restrict__ W, _Float16* __restrict__ frags)
{
    constexpr int KP = (K + 31) & ~31, KT = KP / 32;
    const int frag = blockIdx.x;
    const int nt = frag / KT, kt = frag % KT;
    const int lane = threadIdx.x;
    const int ncol = nt * 16 + (lane & 15);
    _Float16* o = frags + ((size_t)frag * 32 + lane) * 16;
    #pragma unroll
    for (int slot = 0; slot < 16; ++slot) {
        const int k = kt * 32 + frag_k_b(lane, slot);
        o[slot] = (_Float16)((k < K && ncol < 50) ? W[(size_t)k * 50 + ncol] : 0.f);
    }
}

// ---------------------------------------------------------------------------
// Layer-1 direct conv: BN1 -> edge-pad -> conv (1->OC) -> 2x2 maxpool -> ReLU
// -> BN2 fold -> f16 map edge-padded by 1.
template<int INH, int INW, int PAD, int KH, int KW, int OC>
__global__ __launch_bounds__(256)
void conv1_kernel(const float* __restrict__ x,
                  const float* __restrict__ g1, const float* __restrict__ b1,
                  const float* __restrict__ w, const float* __restrict__ cb,
                  const float* __restrict__ g2, const float* __restrict__ b2,
                  _Float16* __restrict__ out)
{
    constexpr int PH = INH + 2 * PAD, PW = INW + 2 * PAD;
    constexpr int CH = PH - KH + 1,  CW = PW - KW + 1;
    constexpr int POH = CH / 2,      POW = CW / 2;
    constexpr int OPH = POH + 2,     OPW = POW + 2;

    __shared__ float sin_[PH * PW];
    __shared__ float sw_[OC * KH * KW];

    const int img = blockIdx.x;
    const float* xin = x + (size_t)img * (INH * INW);
    const float s1 = g1[0] * rsqrtf(1.0f + BN_EPS);
    const float bb1 = b1[0];

    for (int i = threadIdx.x; i < PH * PW; i += 256) {
        int y = i / PW - PAD, xx = i % PW - PAD;
        y  = min(max(y, 0), INH - 1);
        xx = min(max(xx, 0), INW - 1);
        sin_[i] = xin[y * INW + xx] * s1 + bb1;
    }
    for (int i = threadIdx.x; i < OC * KH * KW; i += 256) sw_[i] = w[i];
    __syncthreads();

    _Float16* o = out + (size_t)img * OC * OPH * OPW;
    for (int idx = threadIdx.x; idx < OC * POH * POW; idx += 256) {
        const int oc  = idx / (POH * POW);
        const int rem = idx % (POH * POW);
        const int py = rem / POW, px = rem % POW;
        float mx = -3.4e38f;
        for (int dy = 0; dy < 2; ++dy)
            for (int dx = 0; dx < 2; ++dx) {
                const int oy = py * 2 + dy, ox = px * 2 + dx;
                float acc = 0.f;
                #pragma unroll
                for (int ky = 0; ky < KH; ++ky)
                    #pragma unroll
                    for (int kx = 0; kx < KW; ++kx)
                        acc += sin_[(oy + ky) * PW + ox + kx] * sw_[oc * KH * KW + ky * KW + kx];
                mx = fmaxf(mx, acc);
            }
        float v = fmaxf(mx + cb[oc], 0.f);
        v = v * (g2[oc] * rsqrtf(1.0f + BN_EPS)) + b2[oc];
        const _Float16 hv = (_Float16)v;
        _Float16* ob = o + (size_t)oc * OPH * OPW;
        const int Y = py + 1, X = px + 1;
        ob[Y * OPW + X] = hv;
        const bool tp = (py == 0), bm = (py == POH - 1), lf = (px == 0), rt = (px == POW - 1);
        if (tp)       ob[X]                         = hv;
        if (bm)       ob[(OPH - 1) * OPW + X]       = hv;
        if (lf)       ob[Y * OPW]                   = hv;
        if (rt)       ob[Y * OPW + OPW - 1]         = hv;
        if (tp && lf) ob[0]                         = hv;
        if (tp && rt) ob[OPW - 1]                   = hv;
        if (bm && lf) ob[(OPH - 1) * OPW]           = hv;
        if (bm && rt) ob[(OPH - 1) * OPW + OPW - 1] = hv;
    }
}

// ---------------------------------------------------------------------------
// Implicit-GEMM WMMA conv, LDS-staged activations (async), prepacked A frags.
// One block per image; each wave owns one N-tile and ALL M-tiles (register
// accumulators), so one B fragment feeds MT WMMAs.
// MODE 0: flat f16 store with per-image row stride OSTRIDE halves.
// MODE 1: BN fold + edge-pad-1 f16 map.
template<int IC, int KH, int KW, int INH, int INW, int OC, int MODE, int OSTRIDE, int NWAVES>
__global__ __launch_bounds__(NWAVES * 32)
void conv_wmma_kernel(const _Float16* __restrict__ in,
                      const _Float16* __restrict__ afr,   // prepacked A frags
                      const float* __restrict__ cb,
                      const float* __restrict__ g, const float* __restrict__ bt,
                      _Float16* __restrict__ out)
{
    constexpr int CH = INH - KH + 1, CW = INW - KW + 1;
    constexpr int POH = CH / 2, POW = CW / 2;
    constexpr int NTOT = 4 * POH * POW;          // CH,CW even in all uses
    constexpr int K  = IC * KH * KW;
    constexpr int KP = (K + 31) & ~31, KT = KP / 32;
    constexpr int MT = (OC + 15) / 16, NT = (NTOT + 15) / 16;
    constexpr int INSZ = IC * INH * INW;         // halves
    constexpr int CHUNKS = INSZ * 2 / 16;        // 16B chunks (all sizes divisible)

    __shared__ _Float16 sIn[INSZ];

    const int img = blockIdx.x;
    const int tid = threadIdx.x;
    const int lane = tid & 31, wave = tid >> 5;

    // --- async stage activations into LDS ---
    {
        const _Float16* gsrc = in + (size_t)img * INSZ;
        for (int i = tid; i < CHUNKS; i += NWAVES * 32)
            async_copy16(gsrc + i * 8, sIn + i * 8);
        async_wait0();
    }
    __syncthreads();

    for (int nt = wave; nt < NT; nt += NWAVES) {
        const int ncol = nt * 16 + (lane & 15);
        const bool nvalid = (ncol < NTOT);
        const int wdw = ncol >> 2, e = ncol & 3;
        const int py = wdw / POW, px = wdw % POW;
        const int oy = py * 2 + (e >> 1), ox = px * 2 + (e & 1);

        CF acc[MT];
        #pragma unroll
        for (int mt = 0; mt < MT; ++mt)
            #pragma unroll
            for (int i = 0; i < 8; ++i) acc[mt].f[i] = 0.f;

        for (int kt = 0; kt < KT; ++kt) {
            AF B;
            #pragma unroll
            for (int slot = 0; slot < 16; ++slot) {
                const int k = kt * 32 + frag_k_b(lane, slot);
                _Float16 bv = (_Float16)0.f;
                if (nvalid && k < K) {
                    const int ic = k / (KH * KW);
                    const int r  = k % (KH * KW);
                    const int ky = r / KW, kx = r % KW;
                    bv = sIn[ic * INH * INW + (oy + ky) * INW + (ox + kx)];
                }
                B.h[slot] = bv;
            }
            #pragma unroll
            for (int mt = 0; mt < MT; ++mt) {
                AF A;
                const uint4* ap = (const uint4*)(afr + ((size_t)(mt * KT + kt) * 32 + lane) * 16);
                A.q[0] = ap[0]; A.q[1] = ap[1];
                acc[mt].v = __builtin_amdgcn_wmma_f32_16x16x32_f16(
                    false, A.v, false, B.v, (short)0, acc[mt].v, false, false);
            }
        }

        // 2x2 maxpool across the 4 columns of each window, bias, ReLU, store
        #pragma unroll
        for (int mt = 0; mt < MT; ++mt) {
            #pragma unroll
            for (int r = 0; r < 8; ++r) {
                float v0 = acc[mt].f[r];
                v0 = fmaxf(v0, __shfl_xor(v0, 1, 32));
                v0 = fmaxf(v0, __shfl_xor(v0, 2, 32));
                const int oc = mt * 16 + r + ((lane & 16) ? 8 : 0);
                if (((lane & 3) == 0) && nvalid && oc < OC) {
                    float val = fmaxf(v0 + cb[oc], 0.f);
                    if (MODE == 0) {
                        out[(size_t)img * OSTRIDE + (size_t)oc * (POH * POW) + wdw] = (_Float16)val;
                    } else {
                        constexpr int OPH = POH + 2, OPW = POW + 2;
                        val = val * (g[oc] * rsqrtf(1.0f + BN_EPS)) + bt[oc];
                        const _Float16 hv = (_Float16)val;
                        _Float16* ob = out + (size_t)img * OC * OPH * OPW + (size_t)oc * OPH * OPW;
                        const int Y = py + 1, X = px + 1;
                        ob[Y * OPW + X] = hv;
                        const bool tp = (py == 0), bm = (py == POH - 1), lf = (px == 0), rt = (px == POW - 1);
                        if (tp)       ob[X]                         = hv;
                        if (bm)       ob[(OPH - 1) * OPW + X]       = hv;
                        if (lf)       ob[Y * OPW]                   = hv;
                        if (rt)       ob[Y * OPW + OPW - 1]         = hv;
                        if (tp && lf) ob[0]                         = hv;
                        if (tp && rt) ob[OPW - 1]                   = hv;
                        if (bm && lf) ob[(OPH - 1) * OPW]           = hv;
                        if (bm && rt) ob[(OPH - 1) * OPW + OPW - 1] = hv;
                    }
                }
            }
        }
    }
}

// ---------------------------------------------------------------------------
// WMMA FC head: out[M][50] = relu( X[M][K](f16, row stride KS) @ W + b ).
// 4 waves per block; each wave = one M-tile, holds all 4 N-tile accumulators.
// A fragment = two contiguous 16B chunks per lane; B frags prepacked.
template<int K, int KS>
__global__ __launch_bounds__(128)
void fc_wmma_kernel(const _Float16* __restrict__ X,
                    const _Float16* __restrict__ bfr,
                    const float* __restrict__ b, float* __restrict__ out)
{
    constexpr int KP = (K + 31) & ~31, KT = KP / 32;
    const int tid = threadIdx.x, lane = tid & 31, wave = tid >> 5;
    const int mt = blockIdx.x * 4 + wave;       // 256 M-tiles total
    const int off = (lane & 16) ? 8 : 0;
    const _Float16* xrow = X + (size_t)(mt * 16 + (lane & 15)) * KS;

    CF acc[4];
    #pragma unroll
    for (int nt = 0; nt < 4; ++nt)
        #pragma unroll
        for (int i = 0; i < 8; ++i) acc[nt].f[i] = 0.f;

    for (int kt = 0; kt < KT; ++kt) {
        const int k0 = kt * 32;
        AF A;
        if (k0 + 32 <= K) {
            A.q[0] = *(const uint4*)(xrow + k0 + off);
            A.q[1] = *(const uint4*)(xrow + k0 + 16 + off);
        } else {
            #pragma unroll
            for (int s = 0; s < 16; ++s) {
                const int k = k0 + frag_k_a(lane, s);
                A.h[s] = (k < K) ? xrow[k] : (_Float16)0.f;
            }
        }
        #pragma unroll
        for (int nt = 0; nt < 4; ++nt) {
            AF B;
            const uint4* bp = (const uint4*)(bfr + ((size_t)(nt * KT + kt) * 32 + lane) * 16);
            B.q[0] = bp[0]; B.q[1] = bp[1];
            acc[nt].v = __builtin_amdgcn_wmma_f32_16x16x32_f16(
                false, A.v, false, B.v, (short)0, acc[nt].v, false, false);
        }
    }
    #pragma unroll
    for (int nt = 0; nt < 4; ++nt) {
        const int ncol = nt * 16 + (lane & 15);
        #pragma unroll
        for (int r = 0; r < 8; ++r) {
            const int row = mt * 16 + r + ((lane & 16) ? 8 : 0);
            if (ncol < 50) out[(size_t)row * 50 + ncol] = fmaxf(acc[nt].f[r] + b[ncol], 0.f);
        }
    }
}

// ---------------------------------------------------------------------------
__global__ __launch_bounds__(32)
void head_kernel(const float* __restrict__ h1, const float* __restrict__ h2,
                 const float* __restrict__ h3,
                 const float* __restrict__ x4, const float* __restrict__ x5,
                 const float* __restrict__ g31, const float* __restrict__ b31,
                 const float* __restrict__ g32, const float* __restrict__ b32,
                 const float* __restrict__ w41, const float* __restrict__ b41,
                 const float* __restrict__ w431, const float* __restrict__ b431,
                 const float* __restrict__ w432, const float* __restrict__ b432,
                 float* __restrict__ yfin)
{
    const int brow = blockIdx.x;
    const int t = threadIdx.x;
    const int row = brow * 32 + t;
    float z0 = b41[0], z1 = b41[1];
    const float* f1 = h1 + (size_t)row * 50;
    const float* f2 = h2 + (size_t)row * 50;
    const float* f3 = h3 + (size_t)row * 50;
    for (int j = 0; j < 50; ++j) { float v = f1[j]; z0 += v * w41[j * 2];         z1 += v * w41[j * 2 + 1]; }
    for (int j = 0; j < 50; ++j) { float v = f2[j]; z0 += v * w41[(50 + j) * 2];  z1 += v * w41[(50 + j) * 2 + 1]; }
    for (int j = 0; j < 50; ++j) { float v = f3[j]; z0 += v * w41[(100 + j) * 2]; z1 += v * w41[(100 + j) * 2 + 1]; }
    const float rs = rsqrtf(1.0f + BN_EPS);
    for (int j = 0; j < 4; ++j) {
        float v = fmaxf(x4[(size_t)row * 4 + j] * g31[j] * rs + b31[j], 0.f);
        z0 += v * w41[(150 + j) * 2]; z1 += v * w41[(150 + j) * 2 + 1];
    }
    {
        float v = fmaxf(x5[row] * g32[0] * rs + b32[0], 0.f);
        z0 += v * w41[154 * 2]; z1 += v * w41[154 * 2 + 1];
    }
    z0 = fmaxf(z0, 0.f); z1 = fmaxf(z1, 0.f);
    const float mz = fmaxf(z0, z1);
    const float e0 = expf(z0 - mz), e1 = expf(z1 - mz);
    const float inv = 1.0f / (e0 + e1);
    float s0 = e0 * inv, s1 = e1 * inv;
    for (int offv = 16; offv >= 1; offv >>= 1) {
        s0 += __shfl_xor(s0, offv, 32);
        s1 += __shfl_xor(s1, offv, 32);
    }
    if (t == 0) {
        const float ys[2] = { s0, s1 };
        for (int c = 0; c < 2; ++c) {
            const float yc = (ys[c] > 0.9f) ? ys[c] : 0.f;
            float a2 = 0.f;
            for (int j = 0; j < 50; ++j) {
                const float hj = fmaxf(yc * w431[c * 50 + j] + b431[c * 50 + j], 0.f);
                a2 += hj * w432[c * 50 + j];
            }
            yfin[brow * 2 + c] = fmaxf(a2 + b432[c], 0.f);
        }
    }
}

__global__ __launch_bounds__(256)
void norm_kernel(const float* __restrict__ yfin, float* __restrict__ out)
{
    __shared__ float red[256];
    const int t = threadIdx.x;
    const float v = yfin[t];
    red[t] = v; __syncthreads();
    for (int s = 128; s > 0; s >>= 1) { if (t < s) red[t] += red[t + s]; __syncthreads(); }
    const float mean = red[0] / 256.0f;
    __syncthreads();
    red[t] = (v - mean) * (v - mean); __syncthreads();
    for (int s = 128; s > 0; s >>= 1) { if (t < s) red[t] += red[t + s]; __syncthreads(); }
    const float sd = sqrtf(red[0] / 255.0f);
    if (t < 128) out[t] = (yfin[2 * t] - mean) / sd;
}

// ---------------------------------------------------------------------------
extern "C" void kernel_launch(void* const* d_in, const int* in_sizes, int n_in,
                              void* d_out, int out_size, void* d_ws, size_t ws_size,
                              hipStream_t stream) {
    const float* x1 = (const float*)d_in[0];
    const float* x2 = (const float*)d_in[1];
    const float* x3 = (const float*)d_in[2];
    const float* x4 = (const float*)d_in[3];
    const float* x5 = (const float*)d_in[4];
    #define PAR(i) ((const float*)d_in[6 + (i)])

    const int m = 4096;
    char* ws = (char*)d_ws;
    size_t off = 0;
    auto take = [&](size_t bytes) -> void* {
        void* p = ws + off;
        off += (bytes + 255) & ~(size_t)255;
        return p;
    };
    _Float16* c11map  = (_Float16*)take((size_t)m * 10 * 22 * 32 * 2);
    _Float16* c12map  = (_Float16*)take((size_t)m * 50 * 6 * 10 * 2);
    _Float16* c13map1 = (_Float16*)take((size_t)m * 70 * 4 * 12 * 2);
    _Float16* c13map2 = (_Float16*)take((size_t)m * 140 * 4 * 6 * 2);
    _Float16* f2buf   = (_Float16*)take((size_t)m * 3008 * 2);   // K=3000, stride 3008
    _Float16* f1buf   = (_Float16*)take((size_t)m * 800 * 2);    // K=800
    _Float16* f3buf   = (_Float16*)take((size_t)m * 416 * 2);    // K=400, stride 416
    float*    h2      = (float*)take((size_t)m * 50 * 4);
    float*    h1      = (float*)take((size_t)m * 50 * 4);
    float*    h3      = (float*)take((size_t)m * 50 * 4);
    float*    yfin    = (float*)take(256 * 4);
    // prepacked weight fragments (1 KB each)
    _Float16* afr11 = (_Float16*)take((size_t)2  * 3  * 1024);   // OC20  K90
    _Float16* afr12 = (_Float16*)take((size_t)7  * 15 * 1024);   // OC100 K450
    _Float16* afr13a = (_Float16*)take((size_t)9  * 11 * 1024);  // OC140 K350
    _Float16* afr13b = (_Float16*)take((size_t)13 * 40 * 1024);  // OC200 K1260
    _Float16* bfr21 = (_Float16*)take((size_t)4 * 94 * 1024);    // K3000
    _Float16* bfr22 = (_Float16*)take((size_t)4 * 25 * 1024);    // K800
    _Float16* bfr23 = (_Float16*)take((size_t)4 * 13 * 1024);    // K400

    // --- weight pre-pack (cheap; deterministic each call) ---
    pack_a_kernel<20, 90>   <<<2 * 3,   32, 0, stream>>>(PAR(6),  afr11);
    pack_a_kernel<100, 450> <<<7 * 15,  32, 0, stream>>>(PAR(14), afr12);
    pack_a_kernel<140, 350> <<<9 * 11,  32, 0, stream>>>(PAR(22), afr13a);
    pack_a_kernel<200, 1260><<<13 * 40, 32, 0, stream>>>(PAR(26), afr13b);
    pack_b_fc_kernel<3000><<<4 * 94, 32, 0, stream>>>(PAR(28), bfr21);
    pack_b_fc_kernel<800> <<<4 * 25, 32, 0, stream>>>(PAR(30), bfr22);
    pack_b_fc_kernel<400> <<<4 * 13, 32, 0, stream>>>(PAR(32), bfr23);

    // --- layer-1 direct convs (1 input channel) ---
    conv1_kernel<40, 60, 2, 5, 5, 10><<<m, 256, 0, stream>>>(
        x2, PAR(0), PAR(1), PAR(2), PAR(3), PAR(4), PAR(5), c11map);
    conv1_kernel<8, 16, 2, 5, 5, 50><<<m, 256, 0, stream>>>(
        x1, PAR(8), PAR(9), PAR(10), PAR(11), PAR(12), PAR(13), c12map);
    conv1_kernel<3, 24, 1, 1, 7, 70><<<m, 256, 0, stream>>>(
        x3, PAR(16), PAR(17), PAR(18), PAR(19), PAR(20), PAR(21), c13map1);

    // --- WMMA implicit-GEMM convs (async LDS staging) ---
    conv_wmma_kernel<10, 3, 3, 22, 32, 20, 0, 3008, 4><<<m, 128, 0, stream>>>(
        c11map, afr11, PAR(7), nullptr, nullptr, f2buf);
    conv_wmma_kernel<50, 3, 3, 6, 10, 100, 0, 800, 2><<<m, 64, 0, stream>>>(
        c12map, afr12, PAR(15), nullptr, nullptr, f1buf);
    conv_wmma_kernel<70, 1, 5, 4, 12, 140, 1, 0, 2><<<m, 64, 0, stream>>>(
        c13map1, afr13a, PAR(23), PAR(24), PAR(25), c13map2);
    conv_wmma_kernel<140, 3, 3, 4, 6, 200, 0, 416, 1><<<m, 32, 0, stream>>>(
        c13map2, afr13b, PAR(27), nullptr, nullptr, f3buf);

    // --- WMMA FC heads (64 blocks x 4 waves = 256 M-tiles) ---
    fc_wmma_kernel<3000, 3008><<<64, 128, 0, stream>>>(f2buf, bfr21, PAR(29), h2);
    fc_wmma_kernel<800, 800>  <<<64, 128, 0, stream>>>(f1buf, bfr22, PAR(31), h1);
    fc_wmma_kernel<400, 416>  <<<64, 128, 0, stream>>>(f3buf, bfr23, PAR(33), h3);

    // --- heads / softmax / sub-batch sum / threshold / c43 / normalize ---
    head_kernel<<<128, 32, 0, stream>>>(h1, h2, h3, x4, x5,
                                        PAR(34), PAR(35), PAR(36), PAR(37),
                                        PAR(38), PAR(39),
                                        PAR(40), PAR(41), PAR(42), PAR(43), yfin);
    norm_kernel<<<1, 256, 0, stream>>>(yfin, (float*)d_out);
    #undef PAR
}
